// DepthAwarePoseGAT_86002425135787
// MI455X (gfx1250) — matrix operand order repaired and checked by reading, not compiled, not run on hardware
//
#include <hip/hip_runtime.h>
#include <hip/hip_bf16.h>
#include <math.h>

// ---------------------------------------------------------------------------
// DepthAwarePoseGAT for gfx1250 (MI455X).
// GEMMs via v_wmma_f32_16x16x32_f16 (f16 A/B, f32 accumulate).
// Edge phase: 3-pass stable global softmax + coalesced atomic scatter-add.
// ---------------------------------------------------------------------------

typedef __attribute__((ext_vector_type(16))) _Float16 v16h;
typedef __attribute__((ext_vector_type(8)))  _Float16 v8h;
typedef __attribute__((ext_vector_type(8)))  float    v8f;

#define NNODES 50000
#define NEDGES 800000
#define ETOT   (NNODES + NEDGES)
#define GAT_NEG_SLOPE 0.2f

// order-preserving float<->uint encoding for atomic max
static __device__ __forceinline__ unsigned enc_f32(float f) {
  unsigned u = __float_as_uint(f);
  return (u & 0x80000000u) ? ~u : (u | 0x80000000u);
}
static __device__ __forceinline__ float dec_f32(unsigned u) {
  unsigned b = (u & 0x80000000u) ? (u & 0x7FFFFFFFu) : ~u;
  return __uint_as_float(b);
}

// ---- input prep: concat(x, emb[jt]) -> f16 activations, row stride 32 -----
__global__ void k_prep_input(const float* __restrict__ x,
                             const int* __restrict__ jt,
                             const float* __restrict__ emb,
                             _Float16* __restrict__ act, int n) {
  int i = blockIdx.x * blockDim.x + threadIdx.x;
  if (i >= n) return;
  _Float16* row = act + (size_t)i * 32;
  int t = jt[i];
#pragma unroll
  for (int f = 0; f < 4; ++f)  row[f]     = (_Float16)x[(size_t)i * 4 + f];
#pragma unroll
  for (int f = 0; f < 16; ++f) row[4 + f] = (_Float16)emb[t * 16 + f];
#pragma unroll
  for (int f = 20; f < 32; ++f) row[f] = (_Float16)0.0f;
}

// ---- weight f32[O,Kin] -> f16[O,Kpad] (zero padded) -----------------------
__global__ void k_conv_w(const float* __restrict__ w, _Float16* __restrict__ wh,
                         int O, int Kin, int Kpad) {
  int i = blockIdx.x * blockDim.x + threadIdx.x;
  if (i >= O * Kpad) return;
  int o = i / Kpad, k = i - o * Kpad;
  wh[i] = (k < Kin) ? (_Float16)w[(size_t)o * Kin + k] : (_Float16)0.0f;
}

// ---- GEMM: D[M,O] = A[M,K](f16,row-major) * B[O,K](f16,row-major)^T -------
// one wave per 16x16 tile; K multiple of 32
__global__ void k_gemm_wmma(const _Float16* __restrict__ A,
                            const _Float16* __restrict__ B,
                            float* __restrict__ D, int M, int K, int O) {
  int gid  = blockIdx.x * blockDim.x + threadIdx.x;
  int wave = gid >> 5;
  int lane = gid & 31;
  int tilesO = O >> 4;
  int tm = wave / tilesO;
  int tn = wave - tm * tilesO;
  if (tm * 16 >= M) return;
  int mlo = lane & 15;   // row (A) / col (B,D) within tile
  int hi  = lane >> 4;   // K half-select
  const _Float16* arow = A + (size_t)(tm * 16 + mlo) * K;
  const _Float16* brow = B + (size_t)(tn * 16 + mlo) * K;
  v8f acc = {};
  for (int k0 = 0; k0 < K; k0 += 32) {
    // A frag: elems 0..7 = A[m][k0+hi*8 ..], elems 8..15 = A[m][k0+16+hi*8 ..]
    v8h alo = *(const v8h*)(arow + k0 + hi * 8);
    v8h ahi = *(const v8h*)(arow + k0 + 16 + hi * 8);
    v16h av;
#pragma unroll
    for (int i = 0; i < 8; ++i) { av[i] = alo[i]; av[i + 8] = ahi[i]; }
    // B frag: elems 0..7 = B[n][k0+hi*8 ..], elems 8..15 = B[n][k0+16+hi*8 ..]
    v8h blo = *(const v8h*)(brow + k0 + hi * 8);
    v8h bhi = *(const v8h*)(brow + k0 + 16 + hi * 8);
    v16h bv;
#pragma unroll
    for (int i = 0; i < 8; ++i) { bv[i] = blo[i]; bv[i + 8] = bhi[i]; }
    acc = __builtin_amdgcn_wmma_f32_16x16x32_f16(
        false, av, false, bv, (short)0, acc, false, false);
  }
  // D layout: VGPR r, lanes 0-15 -> M=r, lanes 16-31 -> M=8+r; N = lane&15
  float* drow = D + (size_t)(tm * 16 + hi * 8) * O + tn * 16 + mlo;
#pragma unroll
  for (int r = 0; r < 8; ++r) drow[(size_t)r * O] = acc[r];
}

// ---- per-node attention scores p = h . a_src, q = h . a_dst ---------------
__global__ void k_scores(const float* __restrict__ h,
                         const float* __restrict__ a_src,
                         const float* __restrict__ a_dst,
                         float* __restrict__ p, float* __restrict__ q,
                         int n, int H, int C) {
  int i = blockIdx.x * blockDim.x + threadIdx.x;
  if (i >= n * H) return;
  int node = i / H, hh = i - node * H;
  const float* hrow = h + (size_t)node * H * C + (size_t)hh * C;
  const float* as = a_src + hh * C;
  const float* ad = a_dst + hh * C;
  float sp = 0.f, sq = 0.f;
  for (int c = 0; c < C; ++c) { float v = hrow[c]; sp += v * as[c]; sq += v * ad[c]; }
  p[i] = sp; q[i] = sq;
}

__global__ void k_init_stats(unsigned* umax, float* fsum) {
  int i = threadIdx.x;
  if (i < 4) { umax[i] = 0u; fsum[i] = 0.f; }
}

// ---- pass 1: leaky-relu scores + global per-head max ----------------------
__global__ void k_edge_alpha(const int* __restrict__ ei,
                             const float* __restrict__ eattr,
                             const float* __restrict__ p,
                             const float* __restrict__ q,
                             const float* __restrict__ We,
                             float* __restrict__ alpha,
                             unsigned* __restrict__ umax, int H) {
  int e = blockIdx.x * blockDim.x + threadIdx.x;
  float loc[4];
#pragma unroll
  for (int hh = 0; hh < 4; ++hh) loc[hh] = -3.0e38f;
  if (e < ETOT) {
    int s, d; float ea[4];
    if (e < NEDGES) {
      s = ei[e]; d = ei[NEDGES + e];
#pragma unroll
      for (int f = 0; f < 4; ++f) ea[f] = eattr[(size_t)e * 4 + f];
    } else {
      s = d = e - NEDGES;
#pragma unroll
      for (int f = 0; f < 4; ++f) ea[f] = 0.f;
    }
    for (int hh = 0; hh < H; ++hh) {
      float v = p[(size_t)d * H + hh] + q[(size_t)s * H + hh];
#pragma unroll
      for (int f = 0; f < 4; ++f) v += ea[f] * We[hh * 4 + f];
      v = v > 0.f ? v : GAT_NEG_SLOPE * v;      // leaky relu
      alpha[(size_t)e * H + hh] = v;
      loc[hh] = v;
    }
  }
#pragma unroll
  for (int hh = 0; hh < 4; ++hh) {
    float m = loc[hh];
    for (int off = 16; off; off >>= 1) m = fmaxf(m, __shfl_xor(m, off, 32));
    if ((threadIdx.x & 31) == 0 && hh < H) atomicMax(&umax[hh], enc_f32(m));
  }
}

// ---- pass 2: exp(a - max), global per-head sum ----------------------------
__global__ void k_edge_expsum(float* __restrict__ alpha,
                              const unsigned* __restrict__ umax,
                              float* __restrict__ fsum, int H) {
  int e = blockIdx.x * blockDim.x + threadIdx.x;
  float loc[4] = {0.f, 0.f, 0.f, 0.f};
  if (e < ETOT) {
    for (int hh = 0; hh < H; ++hh) {
      float mx = dec_f32(umax[hh]);
      float v = __expf(alpha[(size_t)e * H + hh] - mx);
      alpha[(size_t)e * H + hh] = v;
      loc[hh] = v;
    }
  }
#pragma unroll
  for (int hh = 0; hh < 4; ++hh) {
    float s = loc[hh];
    for (int off = 16; off; off >>= 1) s += __shfl_xor(s, off, 32);
    if ((threadIdx.x & 31) == 0 && hh < H) atomicAdd(&fsum[hh], s);
  }
}

__global__ void k_recip(const float* __restrict__ fsum, float* __restrict__ finv, int H) {
  int i = threadIdx.x;
  if (i < H) finv[i] = 1.0f / fsum[i];
}

__global__ void k_zero(float* __restrict__ ptr, int n) {
  int i = blockIdx.x * blockDim.x + threadIdx.x;
  if (i < n) ptr[i] = 0.f;
}

// ---- scatter: agg[dst] += h[src] * (alpha/sum), 1 thread per element ------
__global__ void k_aggregate(const int* __restrict__ ei,
                            const float* __restrict__ h,
                            const float* __restrict__ alpha,
                            const float* __restrict__ finv,
                            float* __restrict__ agg, int H, int C) {
  int HC = H * C;
  int t = blockIdx.x * blockDim.x + threadIdx.x;   // < ETOT*HC <= 217.6M
  if (t >= ETOT * HC) return;
  int e = t / HC;
  int idx = t - e * HC;
  int s, d;
  if (e < NEDGES) { s = ei[e]; d = ei[NEDGES + e]; } else { s = d = e - NEDGES; }
  int hh = idx / C;
  float w = alpha[(size_t)e * H + hh] * finv[hh];
  float v = h[(size_t)s * HC + idx] * w;
  atomicAdd(&agg[(size_t)d * HC + idx], v);
}

// ---- head-mean + ELU -> next layer f16 activations (stride 64) ------------
__global__ void k_finalize_mid(const float* __restrict__ agg,
                               _Float16* __restrict__ nxt,
                               int n, int H, int C) {
  int i = blockIdx.x * blockDim.x + threadIdx.x;
  if (i >= n * C) return;
  int node = i / C, c = i - node * C;
  float s = 0.f;
  for (int hh = 0; hh < H; ++hh) s += agg[(size_t)node * H * C + (size_t)hh * C + c];
  s *= (1.0f / (float)H);
  s = s > 0.f ? s : (__expf(s) - 1.0f);            // ELU
  nxt[(size_t)node * 64 + c] = (_Float16)s;
}

__global__ void k_finalize_last(const float* __restrict__ agg,
                                float* __restrict__ out, int n) {
  int i = blockIdx.x * blockDim.x + threadIdx.x;
  if (i < n) out[i] = agg[i];                      // H==1: mean is identity
}

// ---------------------------------------------------------------------------
extern "C" void kernel_launch(void* const* d_in, const int* in_sizes, int n_in,
                              void* d_out, int out_size, void* d_ws, size_t ws_size,
                              hipStream_t stream) {
  (void)in_sizes; (void)n_in; (void)out_size; (void)ws_size;
  const float* x      = (const float*)d_in[0];
  const int*   ei     = (const int*)d_in[1];
  const float* eattr  = (const float*)d_in[2];
  const int*   jt     = (const int*)d_in[3];
  const float* emb    = (const float*)d_in[4];
  const float* W0     = (const float*)d_in[5];
  const float* as0    = (const float*)d_in[6];
  const float* ad0    = (const float*)d_in[7];
  const float* We0    = (const float*)d_in[8];
  const float* W1     = (const float*)d_in[9];
  const float* as1    = (const float*)d_in[10];
  const float* ad1    = (const float*)d_in[11];
  const float* We1    = (const float*)d_in[12];
  const float* W2     = (const float*)d_in[13];
  const float* as2    = (const float*)d_in[14];
  const float* ad2    = (const float*)d_in[15];
  const float* We2    = (const float*)d_in[16];

  char* ws = (char*)d_ws;
  size_t off = 0;
  auto take = [&](size_t bytes) -> void* {
    void* ptr = (void*)(ws + off);
    off += (bytes + 255) & ~(size_t)255;
    return ptr;
  };
  _Float16* actf16 = (_Float16*)take((size_t)NNODES * 64 * sizeof(_Float16));
  _Float16* wf16   = (_Float16*)take((size_t)256 * 64 * sizeof(_Float16));
  float*    hbuf   = (float*)take((size_t)NNODES * 256 * sizeof(float));
  float*    agg    = (float*)take((size_t)NNODES * 256 * sizeof(float));
  float*    pbuf   = (float*)take((size_t)NNODES * 4 * sizeof(float));
  float*    qbuf   = (float*)take((size_t)NNODES * 4 * sizeof(float));
  float*    alpha  = (float*)take((size_t)ETOT * 4 * sizeof(float));
  unsigned* umax   = (unsigned*)take(4 * sizeof(unsigned));
  float*    fsum   = (float*)take(4 * sizeof(float));
  float*    finv   = (float*)take(4 * sizeof(float));

  auto run_layer = [&](const float* W, const float* asrc, const float* adst,
                       const float* We, int Kin, int Kpad, int H, int C, bool last) {
    int O = H * C;
    int nw = O * Kpad;
    k_conv_w<<<(nw + 255) / 256, 256, 0, stream>>>(W, wf16, O, Kin, Kpad);
    int waves = (NNODES / 16) * (O / 16);
    k_gemm_wmma<<<(waves * 32 + 255) / 256, 256, 0, stream>>>(actf16, wf16, hbuf,
                                                              NNODES, Kpad, O);
    k_scores<<<(NNODES * H + 255) / 256, 256, 0, stream>>>(hbuf, asrc, adst,
                                                           pbuf, qbuf, NNODES, H, C);
    k_init_stats<<<1, 32, 0, stream>>>(umax, fsum);
    k_edge_alpha<<<(ETOT + 255) / 256, 256, 0, stream>>>(ei, eattr, pbuf, qbuf,
                                                         We, alpha, umax, H);
    k_edge_expsum<<<(ETOT + 255) / 256, 256, 0, stream>>>(alpha, umax, fsum, H);
    k_recip<<<1, 32, 0, stream>>>(fsum, finv, H);
    int nelem = NNODES * O;
    k_zero<<<(nelem + 255) / 256, 256, 0, stream>>>(agg, nelem);
    long long tot = (long long)ETOT * O;
    k_aggregate<<<(unsigned)((tot + 255) / 256), 256, 0, stream>>>(ei, hbuf, alpha,
                                                                   finv, agg, H, C);
    if (!last)
      k_finalize_mid<<<(NNODES * C + 255) / 256, 256, 0, stream>>>(agg, actf16,
                                                                   NNODES, H, C);
    else
      k_finalize_last<<<(NNODES * C + 255) / 256, 256, 0, stream>>>(agg, (float*)d_out,
                                                                    NNODES * C);
  };

  k_prep_input<<<(NNODES + 255) / 256, 256, 0, stream>>>(x, jt, emb, actf16, NNODES);
  run_layer(W0, as0, ad0, We0, /*Kin*/20, /*Kpad*/32, /*H*/4, /*C*/64,  false);
  run_layer(W1, as1, ad1, We1, /*Kin*/64, /*Kpad*/64, /*H*/4, /*C*/64,  false);
  run_layer(W2, as2, ad2, We2, /*Kin*/64, /*Kpad*/64, /*H*/1, /*C*/128, true);
}
